// DNRI_MLP_Decoder_55198919688338
// MI455X (gfx1250) — matrix operand order repaired and checked by reading, not compiled, not run on hardware
//
#include <hip/hip_runtime.h>

typedef __attribute__((ext_vector_type(16))) _Float16 v16h;
typedef __attribute__((ext_vector_type(8)))  _Float16 h8v;
typedef __attribute__((ext_vector_type(8)))  float    v8f;

#define B_ 512
#define V_ 64
#define D_ 32
#define H_ 64
#define A_ 8
#define E_ 4032          // V*(V-1)
#define NT_ (E_ / 16)    // 252 edge tiles of 16
#define HALF_LOG_2PI 0.9189385332046727f
#define LOG2F_ 0.6931471805599453f

// single-instruction relu (no canonicalize; relu(NaN) -> 0 acceptable here)
__device__ __forceinline__ float relu_fast(float x) {
  float y;
  asm("v_max_num_f32 %0, 0, %1" : "=v"(y) : "v"(x));
  return y;
}

// Phase 1 scratch: edge MLP
struct Phase1 {
  _Float16 xr[V_ * H_];        // f16(inputs @ m1_w[:D] + m1_b)  (recv part)
  _Float16 xs[V_ * H_];        // f16(inputs @ m1_w[D:])         (send part)
  float    wstage[2 * D_ * H_];// m1_w staging
  _Float16 w2h[H_ * H_];       // m2_w in f16 (WMMA B matrix)
  float    we[E_];             // edges[b,:,1]
  int      recv[E_];
  int      send[E_];
  float    b2[H_];             // m2_b
};

// Phase 2 scratch: policy head
struct Phase2 {
  float w1[(D_ + H_) * H_];
  float w2[H_ * H_];
  float h1[V_ * H_];
  float h2[V_ * H_];
  float muw[H_ * A_];
  float lsw[H_ * A_];
  float b1[H_];
  float b2f[H_];
  float mub[A_];
  float lsb[A_];
};

__global__ __launch_bounds__(256)
void dnri_fused(const float* __restrict__ inputs, const float* __restrict__ edges,
                const float* __restrict__ m1_w,  const float* __restrict__ m1_b,
                const float* __restrict__ m2_w,  const float* __restrict__ m2_b,
                const float* __restrict__ fc1_w, const float* __restrict__ fc1_b,
                const float* __restrict__ fc2_w, const float* __restrict__ fc2_b,
                const float* __restrict__ mu_w,  const float* __restrict__ mu_b,
                const float* __restrict__ ls_w,  const float* __restrict__ ls_b,
                const int* __restrict__ send_e,  const int* __restrict__ recv_e,
                float* __restrict__ out)
{
  __shared__ float sIn[V_ * D_];   // inputs[b]           (persists both phases)
  __shared__ float sAgg[V_ * H_];  // scatter-sum target  (persists both phases)
  __shared__ union { Phase1 p1; Phase2 p2; } u;

  const int tid = threadIdx.x;
  const int b   = blockIdx.x;

  // ---------- Phase 0: stage per-batch inputs + layer-1 weights ----------
  for (int i = tid; i < V_ * D_; i += 256) sIn[i] = inputs[(size_t)b * V_ * D_ + i];
  for (int i = tid; i < 2 * D_ * H_; i += 256) u.p1.wstage[i] = m1_w[i];
  for (int i = tid; i < V_ * H_; i += 256) sAgg[i] = 0.f;
  __syncthreads();

  // ---------- Layer-1 factored through nodes: Xr = x@W[:D]+b1, Xs = x@W[D:] ----------
  for (int idx = tid; idx < V_ * H_; idx += 256) {
    const int v = idx >> 6, j = idx & 63;
    float ar = m1_b[j];   // fold bias into recv half
    float as = 0.f;
    #pragma unroll
    for (int d = 0; d < D_; ++d) {
      const float x = sIn[v * D_ + d];
      ar += x * u.p1.wstage[d * H_ + j];
      as += x * u.p1.wstage[(D_ + d) * H_ + j];
    }
    u.p1.xr[idx] = (_Float16)ar;
    u.p1.xs[idx] = (_Float16)as;
  }
  // stage layer-2 weights (f16), per-batch edge gates (coalesced f2), index lists
  for (int i = tid; i < H_ * H_; i += 256) u.p1.w2h[i] = (_Float16)m2_w[i];
  {
    const float2* eg = (const float2*)(edges + (size_t)b * E_ * 2);
    for (int i = tid; i < E_; i += 256) {
      u.p1.we[i]   = eg[i].y;
      u.p1.recv[i] = recv_e[i];
      u.p1.send[i] = send_e[i];
    }
  }
  for (int i = tid; i < H_; i += 256) u.p1.b2[i] = m2_b[i];
  __syncthreads();

  // ---------- Phase 1: per-edge GEMM msg = relu(h @ m2_w + b2) * w_e via WMMA ----------
  const int lane = tid & 31;
  const int wv   = tid >> 5;        // 8 waves
  const int ln15 = lane & 15;
  const int hs   = lane >> 4;       // half-select (0: lanes 0-15, 1: lanes 16-31)

  // B fragments: m2_w tile [32 x 16], lane holds col = lane&15, K = (i>>3)*16 + hs*8 + (i&7)
  v16h bf[4][2];
  #pragma unroll
  for (int n = 0; n < 4; ++n)
    #pragma unroll
    for (int kb = 0; kb < 2; ++kb)
      #pragma unroll
      for (int i = 0; i < 16; ++i) {
        const int k   = kb * 32 + (i >> 3) * 16 + hs * 8 + (i & 7);
        const int col = n * 16 + ln15;
        bf[n][kb][i] = u.p1.w2h[k * H_ + col];
      }
  // m2_b folded into the WMMA accumulator start value (per-column broadcast)
  v8f cinit[4];
  #pragma unroll
  for (int n = 0; n < 4; ++n) {
    const float bb = u.p1.b2[n * 16 + ln15];
    #pragma unroll
    for (int r = 0; r < 8; ++r) cinit[n][r] = bb;
  }
  const h8v hz8 = {(_Float16)0.f, (_Float16)0.f, (_Float16)0.f, (_Float16)0.f,
                   (_Float16)0.f, (_Float16)0.f, (_Float16)0.f, (_Float16)0.f};

  for (int t = wv; t < NT_; t += 8) {
    const int ebase = t * 16;
    const int e  = ebase + ln15;
    const int rv = u.p1.recv[e];
    const int sd = u.p1.send[e];
    // A fragments: h row = edge, 16x32 f16 layout.
    // Each lane's halves form 16B-contiguous runs -> ds_load_b128 + packed f16 math.
    const h8v* xr8 = (const h8v*)(u.p1.xr + rv * H_);   // 8 runs of 8 halves
    const h8v* xs8 = (const h8v*)(u.p1.xs + sd * H_);
    v16h af[2];
    #pragma unroll
    for (int kb = 0; kb < 2; ++kb) {
      const int i0 = kb * 4 + hs;          // run index: k = kb*32 + grp*16 + hs*8
      h8v s0 = xr8[i0]     + xs8[i0];      // grp 0   (v_pk_add_f16 x4)
      h8v s1 = xr8[i0 + 2] + xs8[i0 + 2];  // grp 16
      s0 = __builtin_elementwise_max(s0, hz8);   // v_pk_max_num_f16 x4
      s1 = __builtin_elementwise_max(s1, hz8);
      af[kb] = __builtin_shufflevector(s0, s1, 0, 1, 2, 3, 4, 5, 6, 7,
                                       8, 9, 10, 11, 12, 13, 14, 15);
    }
    // per-row scatter metadata: 8 contiguous ints/floats -> b128 LDS loads
    const int4*   rc = (const int4*)  (u.p1.recv + ebase + 8 * hs);
    const float4* wc = (const float4*)(u.p1.we   + ebase + 8 * hs);
    const int4   r0 = rc[0], r1 = rc[1];
    const float4 w0 = wc[0], w1 = wc[1];
    const int   rRow[8] = {r0.x, r0.y, r0.z, r0.w, r1.x, r1.y, r1.z, r1.w};
    const float wRow[8] = {w0.x, w0.y, w0.z, w0.w, w1.x, w1.y, w1.z, w1.w};

    v8f c[4];
    #pragma unroll
    for (int n = 0; n < 4; ++n) {
      c[n] = cinit[n];
      c[n] = __builtin_amdgcn_wmma_f32_16x16x32_f16(false, af[0], false, bf[n][0],
                                                    (short)0, c[n], false, false);
      c[n] = __builtin_amdgcn_wmma_f32_16x16x32_f16(false, af[1], false, bf[n][1],
                                                    (short)0, c[n], false, false);
    }
    // r-outer / n-inner: one LDS base address per row, n*64B as DS immediate offset
    #pragma unroll
    for (int r = 0; r < 8; ++r) {
      float* p = &sAgg[rRow[r] * H_ + ln15];
      const float w = wRow[r];
      #pragma unroll
      for (int n = 0; n < 4; ++n) {
        atomicAdd(p + n * 16, relu_fast(c[n][r]) * w);   // ds_add_f32 offset:n*64
      }
    }
  }
  __syncthreads();

  // ---------- Phase 2: aug write-out + policy head (0.7 GFLOP total, VALU) ----------
  for (int i = tid; i < (D_ + H_) * H_; i += 256) u.p2.w1[i] = fc1_w[i];
  for (int i = tid; i < H_ * H_; i += 256)        u.p2.w2[i] = fc2_w[i];
  for (int i = tid; i < H_ * A_; i += 256) { u.p2.muw[i] = mu_w[i]; u.p2.lsw[i] = ls_w[i]; }
  for (int i = tid; i < H_; i += 256)      { u.p2.b1[i] = fc1_b[i]; u.p2.b2f[i] = fc2_b[i]; }
  if (tid < A_) { u.p2.mub[tid] = mu_b[tid]; u.p2.lsb[tid] = ls_b[tid]; }

  // aug = [inputs | agg]  -> third output
  const size_t AUG_OFF = (size_t)B_ * V_ * A_ + (size_t)B_ * V_;
  float* aug_out = out + AUG_OFF + (size_t)b * V_ * (D_ + H_);
  for (int i = tid; i < V_ * (D_ + H_); i += 256) {
    const int v = i / (D_ + H_), k = i % (D_ + H_);
    aug_out[i] = (k < D_) ? sIn[v * D_ + k] : sAgg[v * H_ + (k - D_)];
  }
  __syncthreads();

  for (int idx = tid; idx < V_ * H_; idx += 256) {   // h1 = relu(aug @ fc1_w + b)
    const int v = idx >> 6, j = idx & 63;
    float acc = u.p2.b1[j];
    #pragma unroll 8
    for (int k = 0; k < D_; ++k) acc += sIn[v * D_ + k] * u.p2.w1[k * H_ + j];
    #pragma unroll 8
    for (int k = 0; k < H_; ++k) acc += sAgg[v * H_ + k] * u.p2.w1[(D_ + k) * H_ + j];
    u.p2.h1[idx] = relu_fast(acc);
  }
  __syncthreads();
  for (int idx = tid; idx < V_ * H_; idx += 256) {   // h2 = relu(h1 @ fc2_w + b)
    const int v = idx >> 6, j = idx & 63;
    float acc = u.p2.b2f[j];
    #pragma unroll 8
    for (int k = 0; k < H_; ++k) acc += u.p2.h1[v * H_ + k] * u.p2.w2[k * H_ + j];
    u.p2.h2[idx] = relu_fast(acc);
  }
  __syncthreads();

  if (tid < V_) {                                    // heads + logp reduction
    const int v = tid;
    float t1 = 0.f, t2 = 0.f;
    float* mu_out = out + ((size_t)b * V_ + v) * A_;
    #pragma unroll
    for (int a = 0; a < A_; ++a) {
      float mu = u.p2.mub[a], ls = u.p2.lsb[a];
      #pragma unroll 8
      for (int k = 0; k < H_; ++k) {
        const float hk = u.p2.h2[v * H_ + k];
        mu += hk * u.p2.muw[k * A_ + a];
        ls += hk * u.p2.lsw[k * A_ + a];
      }
      ls = fminf(fmaxf(ls, -3.f), 1.f);
      t1 += -ls - HALF_LOG_2PI;
      const float x  = -2.f * mu;
      const float sp = fmaxf(x, 0.f) + log1pf(expf(-fabsf(x)));  // softplus(-2mu)
      t2 += 2.f * (LOG2F_ - mu - sp);
      mu_out[a] = tanhf(mu);
    }
    out[(size_t)B_ * V_ * A_ + (size_t)b * V_ + v] = t1 - t2;
  }
}

extern "C" void kernel_launch(void* const* d_in, const int* in_sizes, int n_in,
                              void* d_out, int out_size, void* d_ws, size_t ws_size,
                              hipStream_t stream) {
  (void)in_sizes; (void)n_in; (void)out_size; (void)d_ws; (void)ws_size;
  const float* inputs = (const float*)d_in[0];
  const float* edges  = (const float*)d_in[1];
  const float* m1_w   = (const float*)d_in[2];
  const float* m1_b   = (const float*)d_in[3];
  const float* m2_w   = (const float*)d_in[4];
  const float* m2_b   = (const float*)d_in[5];
  const float* fc1_w  = (const float*)d_in[6];
  const float* fc1_b  = (const float*)d_in[7];
  const float* fc2_w  = (const float*)d_in[8];
  const float* fc2_b  = (const float*)d_in[9];
  const float* mu_w   = (const float*)d_in[10];
  const float* mu_b   = (const float*)d_in[11];
  const float* ls_w   = (const float*)d_in[12];
  const float* ls_b   = (const float*)d_in[13];
  const int*   send_e = (const int*)d_in[14];
  const int*   recv_e = (const int*)d_in[15];
  dnri_fused<<<dim3(B_), dim3(256), 0, stream>>>(
      inputs, edges, m1_w, m1_b, m2_w, m2_b, fc1_w, fc1_b, fc2_w, fc2_b,
      mu_w, mu_b, ls_w, ls_b, send_e, recv_e, (float*)d_out);
}
